// KalmanLSTMPredictor_27496380629557
// MI455X (gfx1250) — compile-verified, matrix operand order checked by reading, not run on hardware
//
#include <hip/hip_runtime.h>
#include <hip/hip_bf16.h>

typedef _Float16 v16h __attribute__((ext_vector_type(16)));
typedef float    v8f  __attribute__((ext_vector_type(8)));

#define FEAT   32
#define NL     3
#define THIST  20
#define LPRED  30
#define KDT    0.1f
#define KHDT   0.005f

// ---- packed-weight workspace layout (halves) ----
#define F_WIN   0          // 2 frags  (Win 24->32, K padded to 32)
#define F_EWIH  2          // 24 frags (3 layers x 8 N-tiles)
#define F_EWHH  26         // 24
#define F_DWIH  50         // 24
#define F_DWHH  74         // 24
#define F_WOUT  98         // 1 frag   (Wout 32->4, N padded to 16)
#define NFRAGS  99
#define FRAG_H  512        // 32 lanes * 16 halves
// f32 bias region follows, at byte offset NFRAGS*FRAG_H*2
#define BIAS_ENC  0
#define BIAS_DEC  384
#define BIAS_WIN  768
#define BIAS_WOUT 800
#define NBIAS     804

#if __has_builtin(__builtin_amdgcn_sched_barrier)
#define SCHED_FENCE() __builtin_amdgcn_sched_barrier(0)
#else
#define SCHED_FENCE() do {} while (0)
#endif

// Launder a uniform 32-bit offset through an SGPR: defeats LICM/CSE of the
// weight-fragment loads across time steps while PRESERVING the global
// address space of wsH (so loads stay global_load_b128 saddr, not flat).
__device__ __forceinline__ int opaque_s(int v) {
    asm volatile("" : "+s"(v));
    return v;
}
// Laundered zero, VGPR: defeats hoisting of LDS bias loads out of the time loop.
__device__ __forceinline__ int opaque_zero() {
    int z = 0;
    asm volatile("" : "+v"(z));
    return z;
}

// ISA 16-bit A/B fragment addressing (halves within a 1KB fragment):
// element (m,k): lane = m + 16*((k>>3)&1), slot = (k>>4)*8 + (k&7)
__device__ __forceinline__ int aOff(int mm, int k) {
    return ((mm + (((k >> 3) & 1) << 4)) << 4) + ((k >> 4) << 3) + (k & 7);
}

// ---- branch-free fast transcendentals (v_exp_f32 / v_rcp_f32 / v_tanh_f32) ----
__device__ __forceinline__ float fexp(float x) {
    return __builtin_amdgcn_exp2f(x * 1.44269504088896f);
}
__device__ __forceinline__ float fsig(float x) {
    return __builtin_amdgcn_rcpf(1.0f + fexp(-x));
}
#if __has_builtin(__builtin_amdgcn_tanhf)
__device__ __forceinline__ float ftanh(float x) { return __builtin_amdgcn_tanhf(x); }
#else
__device__ __forceinline__ float ftanh(float x) {
    return 1.0f - 2.0f * __builtin_amdgcn_rcpf(fexp(2.0f * x) + 1.0f);
}
#endif

__device__ __forceinline__ void kpred3(float* X, float* P) {
    X[0] += KDT * X[1] + KHDT * X[2];
    X[1] += KDT * X[2];
    float A[9];
#pragma unroll
    for (int j = 0; j < 3; ++j) {
        A[0 * 3 + j] = P[0 * 3 + j] + KDT * P[1 * 3 + j] + KHDT * P[2 * 3 + j];
        A[1 * 3 + j] = P[1 * 3 + j] + KDT * P[2 * 3 + j];
        A[2 * 3 + j] = P[2 * 3 + j];
    }
#pragma unroll
    for (int i = 0; i < 3; ++i) {
        P[i * 3 + 0] = A[i * 3 + 0] + KDT * A[i * 3 + 1] + KHDT * A[i * 3 + 2];
        P[i * 3 + 1] = A[i * 3 + 1] + KDT * A[i * 3 + 2];
        P[i * 3 + 2] = A[i * 3 + 2];
    }
}

// one 16x16 output tile: D = A(16x32) * Bfrag(base + nt) + bias
__device__ __forceinline__ v8f wmma_bias(const _Float16* __restrict__ base, int nt, int lane,
                                         v16h a, float bias) {
    v16h b = *(const v16h*)(base + nt * FRAG_H + lane * 16);
    v8f c = {bias, bias, bias, bias, bias, bias, bias, bias};
    return __builtin_amdgcn_wmma_f32_16x16x32_f16(false, a, false, b, (short)0, c, false, false);
}
// gate tile: bias + x@Wih + h@Whh
__device__ __forceinline__ v8f gate_acc(const _Float16* __restrict__ wihB,
                                        const _Float16* __restrict__ whhB, int nt, int lane,
                                        v16h ax, v16h ah, float bias) {
    v8f c = wmma_bias(wihB, nt, lane, ax, bias);
    v16h bh = *(const v16h*)(whhB + nt * FRAG_H + lane * 16);
    return __builtin_amdgcn_wmma_f32_16x16x32_f16(false, ah, false, bh, (short)0, c, false, false);
}

// ---------------- weight pre-swizzle kernels ----------------
__global__ void pack_weights(const float* __restrict__ WinW,
                             const float* __restrict__ eWih, const float* __restrict__ eWhh,
                             const float* __restrict__ dWih, const float* __restrict__ dWhh,
                             const float* __restrict__ WoutW, _Float16* __restrict__ wsH) {
    int gid = blockIdx.x * blockDim.x + threadIdx.x;
    if (gid >= NFRAGS * FRAG_H) return;
    int fid  = gid / FRAG_H;
    int e    = gid % FRAG_H;
    int lane = e >> 4, idx = e & 15;
    int n  = lane & 15, g1 = lane >> 4;
    int k  = ((idx >> 3) << 4) + (g1 << 3) + (idx & 7);   // inverse of aOff
    float v = 0.f;
    if (fid < 2)        { if (k < 24) v = WinW[(fid * 16 + n) * 24 + k]; }
    else if (fid < 26)  { int q = fid - 2;  v = eWih[(q >> 3) * 4096 + (((q & 7) * 16 + n) << 5) + k]; }
    else if (fid < 50)  { int q = fid - 26; v = eWhh[(q >> 3) * 4096 + (((q & 7) * 16 + n) << 5) + k]; }
    else if (fid < 74)  { int q = fid - 50; v = dWih[(q >> 3) * 4096 + (((q & 7) * 16 + n) << 5) + k]; }
    else if (fid < 98)  { int q = fid - 74; v = dWhh[(q >> 3) * 4096 + (((q & 7) * 16 + n) << 5) + k]; }
    else                { if (n < 4) v = WoutW[n * 32 + k]; }
    wsH[gid] = (_Float16)v;
}

__global__ void pack_bias(const float* __restrict__ ebih, const float* __restrict__ ebhh,
                          const float* __restrict__ dbih, const float* __restrict__ dbhh,
                          const float* __restrict__ Winb, const float* __restrict__ Woutb,
                          float* __restrict__ wsF) {
    int i = blockIdx.x * blockDim.x + threadIdx.x;
    if (i >= NBIAS) return;
    float v;
    if (i < 384)      v = ebih[i] + ebhh[i];
    else if (i < 768) { int q = i - 384; v = dbih[q] + dbhh[q]; }
    else if (i < 800) v = Winb[i - 768];
    else              v = Woutb[i - 800];
    wsF[i] = v;
}

// ---------------- main recurrent kernel: 1 wave == 16 batch rows ----------------
__global__ void __launch_bounds__(256)
klstm_main(const float* __restrict__ hist,
           const float* __restrict__ maxax, const float* __restrict__ maxay,
           const float* __restrict__ vstd,  const float* __restrict__ astd,
           const float* __restrict__ Rx_,   const float* __restrict__ Ry_,
           const float* __restrict__ Gx,    const float* __restrict__ Gy,
           const _Float16* __restrict__ wsH, const float* __restrict__ wsF,
           const int* __restrict__ lenp, float* __restrict__ out, int B) {
    __shared__ _Float16 sh[8][6][512];   // per-wave: [0]=XP frag, [1..2]=x ping-pong, [3..5]=h frags
    __shared__ float    shp[8][64];      // per-wave decoder-head staging
    __shared__ float    sbias[NBIAS];    // step-invariant biases, staged once

    for (int i = threadIdx.x; i < NBIAS; i += 256) sbias[i] = wsF[i];
    __syncthreads();

    const int wv   = threadIdx.x >> 5;
    const int lane = threadIdx.x & 31;
    const int tile = blockIdx.x * 8 + wv;
    const int m0   = tile * 16;
    if (m0 >= B) return;                 // wave-uniform
    const int m    = lane & 15;          // batch row within tile (Kalman lane role)
    const int filt = lane >> 4;          // 0 = x-filter, 1 = y-filter
    const int nn   = lane & 15;          // C/D-layout column
    const int g1m  = lane >> 4;          // C/D-layout row-half selector

    _Float16* fXP   = sh[wv][0];
    _Float16* fX[2] = { sh[wv][1], sh[wv][2] };
    _Float16* fH[3] = { sh[wv][3], sh[wv][4], sh[wv][5] };
    float*    pbuf  = shp[wv];

    // per-lane scatter base: element (v+8*g1m, t*16+nn) -> gbase + t*8 + v*16 (halves)
    const int gbase = (8 * g1m + 16 * (nn >> 3)) * 16 + (nn & 7);
    _Float16* xpA = fXP + m * 16;

    // zero h fragments; zero XP padding columns 24..31 (once)
#pragma unroll
    for (int i = 0; i < 16; ++i) {
        fH[0][lane * 16 + i] = (_Float16)0.f;
        fH[1][lane * 16 + i] = (_Float16)0.f;
        fH[2][lane * 16 + i] = (_Float16)0.f;
    }
    if (lane < 16) {
#pragma unroll
        for (int k = 24; k < 32; ++k) fXP[aOff(lane, k)] = (_Float16)0.f;
    }

    // per-lane Kalman constants
    const float Rxv = Rx_[0] * Rx_[0];
    const float Ryv = Ry_[0] * Ry_[0];
    const float Rl  = filt ? Ryv : Rxv;
    float G[3];
    { const float* Gp = filt ? Gy : Gx; G[0] = Gp[0]; G[1] = Gp[1]; G[2] = Gp[2]; }
    const float amv = filt ? maxay[0] : maxax[0];
    const float am2 = amv * amv;
    float oo[9];
#pragma unroll
    for (int i = 0; i < 3; ++i)
#pragma unroll
        for (int j = 0; j < 3; ++j) oo[i * 3 + j] = G[i] * G[j];

    // kinit (reference uses R_x in diag for BOTH filters)
    float X[3] = { hist[(m0 + m) * THIST * 2 + filt], 0.f, 0.f };
    float P[9] = { Rxv, 0.f, 0.f, 0.f, vstd[0] * vstd[0], 0.f, 0.f, 0.f, astd[0] * astd[0] };

    v8f C[NL][2];
#pragma unroll
    for (int j = 0; j < NL; ++j)
#pragma unroll
        for (int p = 0; p < 2; ++p) C[j][p] = (v8f){0.f,0.f,0.f,0.f,0.f,0.f,0.f,0.f};

    const int LEN = lenp[0];

    for (int phase = 0; phase < 2; ++phase) {
        const int wihF  = phase ? F_DWIH : F_EWIH;
        const int whhF  = phase ? F_DWHH : F_EWHH;
        const int bOff  = phase ? BIAS_DEC : BIAS_ENC;
        const int steps = phase ? LEN : (THIST - 1);

        for (int t = 0; t < steps; ++t) {
            SCHED_FENCE();                 // no cross-step pipelining
            const int bz = opaque_zero();  // defeats hoisting of per-step bias loads
            // per-step laundered table bases: loads stay in-loop (no LICM into
            // VGPRs) yet remain freely schedulable within the step
            const _Float16* winB = wsH + opaque_s(F_WIN * FRAG_H);
            const _Float16* wihB = wsH + opaque_s(wihF * FRAG_H);
            const _Float16* whhB = wsH + opaque_s(whhF * FRAG_H);

            // ---- scatter Kalman state into XP A-fragment (immediate offsets) ----
            if (filt == 0) {   // Xx -> cols 0..2 ; Px -> cols 6..14
                xpA[0] = (_Float16)X[0]; xpA[1] = (_Float16)X[1]; xpA[2] = (_Float16)X[2];
                xpA[6] = (_Float16)P[0]; xpA[7] = (_Float16)P[1];
                xpA[256 + 0] = (_Float16)P[2]; xpA[256 + 1] = (_Float16)P[3];
                xpA[256 + 2] = (_Float16)P[4]; xpA[256 + 3] = (_Float16)P[5];
                xpA[256 + 4] = (_Float16)P[6]; xpA[256 + 5] = (_Float16)P[7];
                xpA[256 + 6] = (_Float16)P[8];
            } else {           // Xy -> cols 3..5 ; Py -> cols 15..23
                xpA[3] = (_Float16)X[0]; xpA[4] = (_Float16)X[1]; xpA[5] = (_Float16)X[2];
                xpA[256 + 7] = (_Float16)P[0];
                xpA[8]  = (_Float16)P[1]; xpA[9]  = (_Float16)P[2]; xpA[10] = (_Float16)P[3];
                xpA[11] = (_Float16)P[4]; xpA[12] = (_Float16)P[5]; xpA[13] = (_Float16)P[6];
                xpA[14] = (_Float16)P[7]; xpA[15] = (_Float16)P[8];
            }
            // ---- x0 = tanh(XP @ Win^T + b) ----
            {
                v16h axp = *(const v16h*)&fXP[lane * 16];
                _Float16* xb = fX[0] + gbase;
#pragma unroll
                for (int nt = 0; nt < 2; ++nt) {
                    v8f d = wmma_bias(winB, nt, lane, axp, sbias[bz + BIAS_WIN + nt * 16 + nn]);
#pragma unroll
                    for (int v = 0; v < 8; ++v)
                        xb[nt * 8 + v * 16] = (_Float16)ftanh(d[v]);
                }
            }
            SCHED_FENCE();
            // ---- LSTM stack (x_{j+1} = c_j per reference) ----
            int cur = 0;
#pragma unroll
            for (int j = 0; j < NL; ++j) {
                v16h ax = *(const v16h*)&fX[cur][lane * 16];
                v16h ah = *(const v16h*)&fH[j][lane * 16];
                const int nxt = cur ^ 1;
                _Float16* hb = fH[j]   + gbase;
                _Float16* xb = fX[nxt] + gbase;
#pragma unroll
                for (int p = 0; p < 2; ++p) {   // feature halves; only 4 accs live
                    v8f gi = gate_acc(wihB, whhB, j*8 + p,     lane, ax, ah, sbias[bz + bOff + j*128 +       p*16 + nn]);
                    v8f gf = gate_acc(wihB, whhB, j*8 + 2 + p, lane, ax, ah, sbias[bz + bOff + j*128 +  32 + p*16 + nn]);
                    v8f gg = gate_acc(wihB, whhB, j*8 + 4 + p, lane, ax, ah, sbias[bz + bOff + j*128 +  64 + p*16 + nn]);
                    v8f go = gate_acc(wihB, whhB, j*8 + 6 + p, lane, ax, ah, sbias[bz + bOff + j*128 +  96 + p*16 + nn]);
#pragma unroll
                    for (int v = 0; v < 8; ++v) {
                        float cn = fsig(gf[v]) * C[j][p][v] + fsig(gi[v]) * ftanh(gg[v]);
                        float hv = fsig(go[v]) * ftanh(cn);
                        C[j][p][v] = cn;
                        hb[p * 8 + v * 16] = (_Float16)hv;
                        xb[p * 8 + v * 16] = (_Float16)cn;
                    }
                }
                SCHED_FENCE();   // cap in-flight weight fragments to ~one layer
                cur = nxt;
            }

            if (phase == 0) {
                // ---- encoder Kalman: predict (+am2*oo) then update with z_{t+1} ----
                kpred3(X, P);
#pragma unroll
                for (int i = 0; i < 9; ++i) P[i] += am2 * oo[i];
                float z = hist[((m0 + m) * THIST + (t + 1)) * 2 + filt];
                float y = z - X[0];
                float S = P[0] + Rl;
                float iS = 1.0f / S;
                float K0 = P[0] * iS, K1 = P[3] * iS, K2 = P[6] * iS;
                X[0] += y * K0; X[1] += y * K1; X[2] += y * K2;
                float r0 = P[0], r1 = P[1], r2 = P[2];
                P[0] -= K0 * r0; P[1] -= K0 * r1; P[2] -= K0 * r2;
                P[3] -= K1 * r0; P[4] -= K1 * r1; P[5] -= K1 * r2;
                P[6] -= K2 * r0; P[7] -= K2 * r1; P[8] -= K2 * r2;
            } else {
                // ---- decoder head: pred = c2 @ Wout^T + b ----
                const _Float16* woutB = wsH + opaque_s(F_WOUT * FRAG_H);
                v16h ac = *(const v16h*)&fX[cur][lane * 16];
                float bb = (nn < 4) ? sbias[bz + BIAS_WOUT + nn] : 0.f;
                v8f d = wmma_bias(woutB, 0, lane, ac, bb);
                if (nn < 4) {
#pragma unroll
                    for (int v = 0; v < 8; ++v) pbuf[(v + 8 * g1m) * 4 + nn] = d[v];
                }
                float p0 = pbuf[m * 4 + filt];          // pred[:,0]/[:,1]
                float pq = pbuf[m * 4 + 2 + filt];      // pred[:,2]/[:,3]
                X[2] = KDT * p0;
                float q2 = pq * pq;
                kpred3(X, P);
#pragma unroll
                for (int i = 0; i < 9; ++i) P[i] += q2 * oo[i];
                float* orow = out + ((size_t)(m0 + m) * LPRED + t) * 5;
                if (filt == 0) { orow[0] = X[0]; orow[2] = sqrtf(P[0]); orow[4] = 0.f; }
                else           { orow[1] = X[0]; orow[3] = sqrtf(P[0]); }
            }
        }
    }
}

extern "C" void kernel_launch(void* const* d_in, const int* in_sizes, int n_in,
                              void* d_out, int out_size, void* d_ws, size_t ws_size,
                              hipStream_t stream) {
    const float* hist  = (const float*)d_in[0];
    const float* maxax = (const float*)d_in[1];
    const float* maxay = (const float*)d_in[2];
    const float* vstd  = (const float*)d_in[3];
    const float* astd  = (const float*)d_in[4];
    const float* Rx_   = (const float*)d_in[5];
    const float* Ry_   = (const float*)d_in[6];
    const float* Gx    = (const float*)d_in[7];
    const float* Gy    = (const float*)d_in[8];
    const float* WinW  = (const float*)d_in[9];
    const float* Winb  = (const float*)d_in[10];
    const float* eWih  = (const float*)d_in[11];
    const float* eWhh  = (const float*)d_in[12];
    const float* ebih  = (const float*)d_in[13];
    const float* ebhh  = (const float*)d_in[14];
    const float* dWih  = (const float*)d_in[15];
    const float* dWhh  = (const float*)d_in[16];
    const float* dbih  = (const float*)d_in[17];
    const float* dbhh  = (const float*)d_in[18];
    const float* WoutW = (const float*)d_in[19];
    const float* Woutb = (const float*)d_in[20];
    const int*   lenp  = (const int*)d_in[21];

    _Float16* wsH = (_Float16*)d_ws;
    float*    wsF = (float*)((char*)d_ws + (size_t)NFRAGS * FRAG_H * sizeof(_Float16));

    pack_weights<<<(NFRAGS * FRAG_H + 255) / 256, 256, 0, stream>>>(
        WinW, eWih, eWhh, dWih, dWhh, WoutW, wsH);
    pack_bias<<<(NBIAS + 255) / 256, 256, 0, stream>>>(
        ebih, ebhh, dbih, dbhh, Winb, Woutb, wsF);

    int B = in_sizes[0] / (THIST * 2);
    int tiles  = B / 16;
    int blocks = (tiles + 7) / 8;
    klstm_main<<<blocks, 256, 0, stream>>>(hist, maxax, maxay, vstd, astd, Rx_, Ry_,
                                           Gx, Gy, wsH, wsF, lenp, (float*)d_out, B);
}
